// TripletLoss_65807488909809
// MI455X (gfx1250) — matrix operand order repaired and checked by reading, not compiled, not run on hardware
//
#include <hip/hip_runtime.h>

// ---------------------------------------------------------------------------
// TripletLoss on MI455X (gfx1250): fused distance-GEMM + masked row-min.
// Precision: fp32 accuracy via bf16 hi/lo split (3x V_WMMA_F32_16X16X32_BF16
// per K=32) -- 2.7x fewer matrix-issue slots than the native f32 16x16x4 path.
// Data path: Tensor Data Mover (tensor_load_to_lds) with double-buffered LDS,
// overlapping DMA of phase p+1 with WMMA of phase p (TENSORcnt + barrier).
// ---------------------------------------------------------------------------

typedef __attribute__((ext_vector_type(16))) __bf16 v16bf;
typedef __attribute__((ext_vector_type(8)))  float  v8f;

#define BQ     8192      // batch (rows of adv / ori)
#define DDIM   1024      // feature dim (K)
#define BM     64        // rows per block
#define BN     128       // columns per N-chunk
#define KC     64        // K slice staged in LDS
#define LDA    (KC + 8)  // LDS row pitch in shorts (16B pad -> stagger banks)
#define NSPLIT 4         // grid.y column partitions
#define NCOLS  (BQ / NSPLIT)
#define NCHUNK (NCOLS / BN)                  // 16 N-chunks per block
#define KPH    (DDIM / KC)                   // 16 K-phases per N-chunk
#define NPHASE (NCHUNK * KPH)                // 256 pipeline phases per block

// LDS layout (units: shorts) within one buffer
#define OFF_AH 0
#define OFF_AL (BM * LDA)
#define OFF_BH (2 * BM * LDA)
#define OFF_BL (2 * BM * LDA + BN * LDA)
#define BUF_SH ((2 * BM + 2 * BN) * LDA)      // 27648 shorts = 55296 B / buffer

#if defined(__gfx1250__) && __has_builtin(__builtin_amdgcn_tensor_load_to_lds) && \
    __has_builtin(__builtin_amdgcn_s_wait_tensorcnt)
#define HAVE_TDM 1
#else
#define HAVE_TDM 0
#endif

struct S4 { unsigned short a, b, c, d; };

__device__ __forceinline__ unsigned short f2bf_rne(float f) {
  unsigned int u = __float_as_uint(f);
  unsigned int r = (u + 0x7FFFu + ((u >> 16) & 1u)) >> 16;  // round-nearest-even
  return (unsigned short)r;
}
__device__ __forceinline__ float bf2f(unsigned short h) {
  return __uint_as_float(((unsigned int)h) << 16);
}

// A fragment (16x32 bf16, MxK): per-lane row, chunks [k0..k0+7], [k0+16..k0+23]
__device__ __forceinline__ v16bf ld_afrag(const unsigned short* rowp, int k0) {
  union { uint4 q[2]; v16bf v; } u;
  u.q[0] = *(const uint4*)(rowp + k0);
  u.q[1] = *(const uint4*)(rowp + k0 + 16);
  return u.v;
}
// B fragment (32x16 bf16, KxN): per-lane column, chunks [k0..k0+7], [k0+16..k0+23]
__device__ __forceinline__ v16bf ld_bfrag(const unsigned short* rowp, int k0) {
  union { uint4 q[2]; v16bf v; } u;
  u.q[0] = *(const uint4*)(rowp + k0);
  u.q[1] = *(const uint4*)(rowp + k0 + 16);
  return u.v;
}

#if HAVE_TDM
typedef unsigned int v4u_ __attribute__((ext_vector_type(4)));
typedef int          v8i_ __attribute__((ext_vector_type(8)));
typedef int          v4i_ __attribute__((ext_vector_type(4)));

// One TDM descriptor: 2D tile (rows x 64 bf16), row = 128B, tensor row stride
// 1024 elems; pads LDS dest by 16B per 128B row -> matches LDA pitch (144B).
__device__ __forceinline__ void tdm_load_2d(unsigned lds_byte_addr,
                                            const unsigned short* gtile,
                                            unsigned rows)
{
  const unsigned long long ga = (unsigned long long)(size_t)gtile;
  v4u_ g0;
  g0.x = 1u;                                                   // count=1, user
  g0.y = lds_byte_addr;                                        // lds_addr
  g0.z = (unsigned)ga;                                         // gaddr[31:0]
  g0.w = ((unsigned)(ga >> 32) & 0x01FFFFFFu) | (2u << 30);    // [56:32]|type=2
  v8i_ g1;
  g1.s0 = (int)((1u << 16)     // data_size = 2 bytes
              | (1u << 20)     // pad_enable
              | (4u << 22)     // pad_interval: 32 DWORDs (=128B row)
              | (3u << 25));   // pad_amount: 4 DWORDs (=16B)
  g1.s1 = (int)((unsigned)DDIM << 16);          // tensor_dim0[15:0] = 1024
  g1.s2 = (int)((unsigned)(2 * BQ) << 16);      // dim0 hi=0 | tensor_dim1 lo
  g1.s3 = (int)((unsigned)KC << 16);            // dim1 hi=0 | tile_dim0 = 64
  g1.s4 = (int)rows;                            // tile_dim1 = rows, tile_dim2=0
  g1.s5 = (int)DDIM;                            // tensor_dim0_stride = 1024
  g1.s6 = 0;
  g1.s7 = 0;
  const v4i_ z4 = {0, 0, 0, 0};
  const v8i_ z8 = {0, 0, 0, 0, 0, 0, 0, 0};
  // 6-arg form (clang-23 / therock-10.0): (g0, g1, g2, g3, g4, cpol)
  __builtin_amdgcn_tensor_load_to_lds(g0, g1, z4, z4, z8, 0);
}

__device__ __forceinline__ void issue_phase(int ph, int row0, int colBase0,
                                            const unsigned short* xhi,
                                            const unsigned short* xlo,
                                            unsigned lds_base_bytes)
{
  const int colBase = colBase0 + (ph >> 4) * BN;
  const int kc = (ph & 15) * KC;
  const size_t aoff = (size_t)(BQ + row0) * DDIM + kc;
  const size_t boff = (size_t)colBase * DDIM + kc;
  tdm_load_2d(lds_base_bytes + OFF_AH * 2u, xhi + aoff, BM);
  tdm_load_2d(lds_base_bytes + OFF_AL * 2u, xlo + aoff, BM);
  tdm_load_2d(lds_base_bytes + OFF_BH * 2u, xhi + boff, BN);
  tdm_load_2d(lds_base_bytes + OFF_BL * 2u, xlo + boff, BN);
}
#endif  // HAVE_TDM

// ---------------------------------------------------------------------------
// Prep: fp32 -> (hi, lo) bf16 planes; exact fp32 row sum-of-squares;
// init min/diag buffers. One block per row (16384 blocks x 256 threads).
// ---------------------------------------------------------------------------
__global__ __launch_bounds__(256)
void prep_kernel(const float* __restrict__ x,
                 unsigned short* __restrict__ xhi,
                 unsigned short* __restrict__ xlo,
                 float* __restrict__ sq,
                 unsigned int* __restrict__ minbits,
                 float* __restrict__ diag)
{
  const int row = blockIdx.x;
  const int tid = threadIdx.x;
  const size_t base = (size_t)row * DDIM + tid * 4;
  const float4 v = *(const float4*)(x + base);
  float xs[4] = {v.x, v.y, v.z, v.w};
  unsigned short h[4], l[4];
  float s = 0.f;
#pragma unroll
  for (int j = 0; j < 4; ++j) {
    h[j] = f2bf_rne(xs[j]);
    l[j] = f2bf_rne(xs[j] - bf2f(h[j]));
    s += xs[j] * xs[j];
  }
  *(S4*)(xhi + base) = S4{h[0], h[1], h[2], h[3]};
  *(S4*)(xlo + base) = S4{l[0], l[1], l[2], l[3]};

#pragma unroll
  for (int off = 16; off > 0; off >>= 1) s += __shfl_down(s, off, 32);
  __shared__ float red[8];
  if ((tid & 31) == 0) red[tid >> 5] = s;
  __syncthreads();
  if (tid == 0) {
    float t = 0.f;
#pragma unroll
    for (int w = 0; w < 8; ++w) t += red[w];
    sq[row] = t;
    if (row < BQ) { minbits[row] = 0x7F800000u; diag[row] = 0.f; }
  }
}

// ---------------------------------------------------------------------------
// Fused GEMM: block = 64 rows x 128-col chunks over 2048 cols; 8 waves as
// 4(M) x 2(N); each wave = one 16-row strip x four 16-col WMMA tiles.
// acc += Ahi*Bhi + Ahi*Blo + Alo*Bhi  (fp32 accumulation)
// ---------------------------------------------------------------------------
__global__ __launch_bounds__(256)
void triplet_gemm(const unsigned short* __restrict__ xhi,
                  const unsigned short* __restrict__ xlo,
                  const float* __restrict__ sq,
                  const int* __restrict__ labels,
                  float* __restrict__ diag,
                  unsigned int* __restrict__ minbits)
{
#if HAVE_TDM
  __shared__ unsigned short smem[2 * BUF_SH];   // double buffer: ~108 KB LDS
#else
  __shared__ unsigned short smem[BUF_SH];
#endif

  const int tid  = threadIdx.x;
  const int lane = tid & 31;
  const int wid  = tid >> 5;      // 0..7
  const int wm   = wid & 3;       // M strip 0..3
  const int wn   = wid >> 2;      // N span 0..1
  const int half = lane >> 4;
  const int l16  = lane & 15;

  const int row0     = blockIdx.x * BM;      // adv row base
  const int colBase0 = blockIdx.y * NCOLS;   // ori col base

  float rmin[8];
  int   labm[8];
  float sqa[8];
#pragma unroll
  for (int r = 0; r < 8; ++r) {
    const int gm = row0 + wm * 16 + half * 8 + r;
    rmin[r] = 3.0e38f;
    labm[r] = labels[gm];
    sqa[r]  = sq[BQ + gm];       // adv norms live in sq[8192..16383]
  }

  const v8f vzero = {0.f, 0.f, 0.f, 0.f, 0.f, 0.f, 0.f, 0.f};

#if HAVE_TDM
  const unsigned lds0 = (unsigned)(size_t)&smem[0];
  if (wid == 0) issue_phase(0, row0, colBase0, xhi, xlo, lds0);
#endif

  int ph = 0;  // flattened pipeline phase counter
  for (int nc = 0; nc < NCHUNK; ++nc) {
    const int colBase = colBase0 + nc * BN;
    v8f acc[4];
#pragma unroll
    for (int t = 0; t < 4; ++t) acc[t] = vzero;

    for (int kci = 0; kci < KPH; ++kci, ++ph) {
#if HAVE_TDM
      const int buf = ph & 1;
      // publish buffer `buf`: DMA complete (wave0) + all waves past last reads
      if (wid == 0) __builtin_amdgcn_s_wait_tensorcnt(0);
      __syncthreads();
      // overlap: DMA phase ph+1 into the other buffer while computing on `buf`
      if (wid == 0 && ph + 1 < NPHASE)
        issue_phase(ph + 1, row0, colBase0, xhi, xlo,
                    lds0 + (unsigned)((buf ^ 1) * BUF_SH) * 2u);
#else
      const int buf = 0;
      const int kc = kci * KC;
      __syncthreads();
#pragma unroll
      for (int i = 0; i < 2; ++i) {
        const int idx = tid + i * 256;
        const int r = idx >> 3;
        const int c = (idx & 7) << 3;
        const size_t g = (size_t)(BQ + row0 + r) * DDIM + (kc + c);
        *(uint4*)&smem[OFF_AH + r * LDA + c] = *(const uint4*)(xhi + g);
        *(uint4*)&smem[OFF_AL + r * LDA + c] = *(const uint4*)(xlo + g);
      }
#pragma unroll
      for (int i = 0; i < 4; ++i) {
        const int idx = tid + i * 256;
        const int n = idx >> 3;
        const int c = (idx & 7) << 3;
        const size_t g = (size_t)(colBase + n) * DDIM + (kc + c);
        *(uint4*)&smem[OFF_BH + n * LDA + c] = *(const uint4*)(xhi + g);
        *(uint4*)&smem[OFF_BL + n * LDA + c] = *(const uint4*)(xlo + g);
        if (kc + KC < DDIM) {
          __builtin_prefetch(xhi + g + KC, 0, 1);
          __builtin_prefetch(xlo + g + KC, 0, 1);
        }
      }
      __syncthreads();
#endif

      // ---- compute on buffer `buf` ----
      const unsigned short* base  = &smem[(size_t)buf * BUF_SH];
      const unsigned short* aHrow = base + OFF_AH + (wm * 16 + l16) * LDA;
      const unsigned short* aLrow = base + OFF_AL + (wm * 16 + l16) * LDA;
#pragma unroll
      for (int ks = 0; ks < KC; ks += 32) {
        const int ka = ks + half * 8;
        const int kb = ks + half * 8;
        v16bf ah = ld_afrag(aHrow, ka);
        v16bf al = ld_afrag(aLrow, ka);
#pragma unroll
        for (int t = 0; t < 4; ++t) {
          const int nrow = wn * 64 + t * 16 + l16;
          v16bf bh = ld_bfrag(base + OFF_BH + nrow * LDA, kb);
          v16bf bl = ld_bfrag(base + OFF_BL + nrow * LDA, kb);
          acc[t] = __builtin_amdgcn_wmma_f32_16x16x32_bf16(
              false, ah, false, bh, (short)0, acc[t], false, false);
          acc[t] = __builtin_amdgcn_wmma_f32_16x16x32_bf16(
              false, ah, false, bl, (short)0, acc[t], false, false);
          acc[t] = __builtin_amdgcn_wmma_f32_16x16x32_bf16(
              false, al, false, bh, (short)0, acc[t], false, false);
        }
      }
    }

    // ---- epilogue once per N-chunk (full K accumulated) ----
#pragma unroll
    for (int t = 0; t < 4; ++t) {
      const int gn = colBase + wn * 64 + t * 16 + l16;
      const int labn = labels[gn];
      const float sqo = sq[gn];
#pragma unroll
      for (int r = 0; r < 8; ++r) {
        const int gm = row0 + wm * 16 + half * 8 + r;
        const float dist = sqa[r] + sqo - 2.0f * acc[t][r];
        if (gm == gn) diag[gm] = dist;            // unique writer per row
        if (labm[r] != labn)
          rmin[r] = fminf(rmin[r], fmaxf(dist, 0.0f));
      }
    }
  }

  // reduce row-min across the 16 lanes sharing each row, then global atomicMin
#pragma unroll
  for (int r = 0; r < 8; ++r) {
    float v = rmin[r];
    v = fminf(v, __shfl_xor(v, 1, 32));
    v = fminf(v, __shfl_xor(v, 2, 32));
    v = fminf(v, __shfl_xor(v, 4, 32));
    v = fminf(v, __shfl_xor(v, 8, 32));
    if (l16 == 0) {
      const int gm = row0 + wm * 16 + half * 8 + r;
      atomicMin(&minbits[gm], __float_as_uint(v));  // valid: v >= 0
    }
  }
}

// ---------------------------------------------------------------------------
// Finish: clip, hinge, mean over 8192 rows (single block)
// ---------------------------------------------------------------------------
__global__ __launch_bounds__(256)
void finish_kernel(const float* __restrict__ diag,
                   const unsigned int* __restrict__ minbits,
                   const float* __restrict__ margin,
                   float* __restrict__ out)
{
  const float m = margin[0];
  float s = 0.f;
  for (int i = threadIdx.x; i < BQ; i += 256) {
    const float c = fminf(fmaxf(diag[i], 1e-12f), 1e12f);
    const float n = fminf(fmaxf(__uint_as_float(minbits[i]), 1e-12f), 1e12f);
    s += fmaxf(c - n + m, 0.f);
  }
#pragma unroll
  for (int off = 16; off > 0; off >>= 1) s += __shfl_down(s, off, 32);
  __shared__ float red[8];
  if ((threadIdx.x & 31) == 0) red[threadIdx.x >> 5] = s;
  __syncthreads();
  if (threadIdx.x == 0) {
    float t = 0.f;
#pragma unroll
    for (int w = 0; w < 8; ++w) t += red[w];
    out[0] = t / (float)BQ;
  }
}

// ---------------------------------------------------------------------------
extern "C" void kernel_launch(void* const* d_in, const int* in_sizes, int n_in,
                              void* d_out, int out_size, void* d_ws, size_t ws_size,
                              hipStream_t stream)
{
  (void)in_sizes; (void)n_in; (void)out_size; (void)ws_size;

  const float* x      = (const float*)d_in[0];   // (16384, 1024) fp32
  const int*   labels = (const int*)d_in[1];     // (8192,)
  const float* margin = (const float*)d_in[2];   // scalar

  // workspace layout (needs ~64.2 MB):
  //   [0,32MB)   xhi bf16   [32MB,64MB) xlo bf16
  //   then sq (16384 f32), minbits (8192 u32), diag (8192 f32)
  char* ws = (char*)d_ws;
  unsigned short* xhi = (unsigned short*)ws;
  unsigned short* xlo = (unsigned short*)(ws + (size_t)32 * 1024 * 1024);
  float*        sq      = (float*)(ws + (size_t)64 * 1024 * 1024);
  unsigned int* minbits = (unsigned int*)(ws + (size_t)64 * 1024 * 1024 + 65536);
  float*        diag    = (float*)(ws + (size_t)64 * 1024 * 1024 + 65536 + 32768);

  hipLaunchKernelGGL(prep_kernel, dim3(2 * BQ), dim3(256), 0, stream,
                     x, xhi, xlo, sq, minbits, diag);
  hipLaunchKernelGGL(triplet_gemm, dim3(BQ / BM, NSPLIT), dim3(256), 0, stream,
                     xhi, xlo, sq, labels, diag, minbits);
  hipLaunchKernelGGL(finish_kernel, dim3(1), dim3(256), 0, stream,
                     diag, minbits, margin, (float*)d_out);
}
